// TransformerGNN_52982716564270
// MI455X (gfx1250) — matrix operand order repaired and checked by reading, not compiled, not run on hardware
//
#include <hip/hip_runtime.h>

#define CH    128
#define HEADS 4
#define HID   32

typedef __attribute__((ext_vector_type(16))) _Float16 v16h;
typedef __attribute__((ext_vector_type(8)))  _Float16 v8h;
typedef __attribute__((ext_vector_type(8)))  float    v8f;

// ---------------------------------------------------------------- utilities

__global__ void fill_kernel(float* __restrict__ p, float v, int n) {
  int i = blockIdx.x * blockDim.x + threadIdx.x;
  if (i < n) p[i] = v;
}

__global__ void f32_to_f16_kernel(const float* __restrict__ s, _Float16* __restrict__ d, int n) {
  int i = blockIdx.x * blockDim.x + threadIdx.x;
  if (i < n) d[i] = (_Float16)s[i];
}

// Fused f32->f16 convert + pack into WMMA B-fragment-major order.
// Packed index: (((kt*8 + nt)*32 + lane)*16 + j)
// Source:       W[(kt*32 + 8*(lane>>4) + (j&7) + 16*(j>>3))*128 + nt*16 + (lane&15)]
__global__ void pack_w_kernel(const float* __restrict__ W, _Float16* __restrict__ P) {
  int idx = blockIdx.x * blockDim.x + threadIdx.x;
  if (idx >= CH * CH) return;
  const int j    = idx & 15;
  const int lane = (idx >> 4) & 31;
  const int frag = idx >> 9;           // kt*8 + nt
  const int kt   = frag >> 3;
  const int nt   = frag & 7;
  const int kk   = kt * 32 + 8 * (lane >> 4) + (j & 7) + 16 * (j >> 3);
  const int nn   = nt * 16 + (lane & 15);
  P[idx] = (_Float16)W[kk * CH + nn];
}

static __device__ __forceinline__ void atomicMaxF(float* addr, float val) {
  // classic monotone-bit-pattern trick; correct for mixed signs
  if (val >= 0.0f) atomicMax((int*)addr, __float_as_int(val));
  else             atomicMin((unsigned int*)addr, __float_as_uint(val));
}

// ---------------------------------------------------------------- WMMA GEMM
// out[nrows,128] = act(A_f16[nrows,128] @ W + bias), W pre-packed fragment-major.
// One wave per 16-row tile; 4 K-steps x 8 N-tiles = 32 v_wmma per wave.
// Packed W (32KB) is staged in LDS once per block and shared by all 8 waves.
__global__ __launch_bounds__(256) void wmma_gemm128(
    const _Float16* __restrict__ A, const _Float16* __restrict__ Wpk,
    const float* __restrict__ bias, float* __restrict__ out,
    int nrows, int do_relu)
{
  __shared__ _Float16 lW[CH * CH];     // 32 KB packed weight

  // cooperative 16B-chunk copy (before any wave can exit)
  {
    const uint4* srcw = (const uint4*)Wpk;
    uint4*       dstw = (uint4*)lW;
    for (int i = threadIdx.x; i < (CH * CH * 2) / 16; i += 256) dstw[i] = srcw[i];
  }
  __syncthreads();

  const int lane   = threadIdx.x & 31;
  const int wave   = threadIdx.x >> 5;
  const int mtiles = (nrows + 15) >> 4;
  const int mtile  = blockIdx.x * 8 + wave;
  if (mtile >= mtiles) return;         // wave-uniform exit: EXEC all-1s at WMMA

  const int kgrp = lane >> 4;          // 0 or 1
  const int ncol = lane & 15;
  int row = mtile * 16 + (lane & 15);
  if (row >= nrows) row = nrows - 1;   // clamp loads (50000 = 3125*16, exact)

  const v8f vzero = {0.f, 0.f, 0.f, 0.f, 0.f, 0.f, 0.f, 0.f};
  v8f acc[8];
#pragma unroll
  for (int i = 0; i < 8; ++i) acc[i] = vzero;

#pragma unroll
  for (int kt = 0; kt < 4; ++kt) {
    // A 16x32 f16 fragment: lane holds K = kt*32 + 8*kgrp + {0..7, 16..23}
    const _Float16* ap = A + (size_t)row * CH + kt * 32 + 8 * kgrp;
    union { v16h v; v8h h[2]; } au;
    au.h[0] = *(const v8h*)(ap);
    au.h[1] = *(const v8h*)(ap + 16);

#pragma unroll
    for (int nt = 0; nt < 8; ++nt) {
      // B fragment: 32 contiguous bytes per lane from LDS-staged packed weight
      const _Float16* bp = lW + (((size_t)(kt * 8 + nt) * 32 + lane) << 4);
      union { v16h v; v8h h[2]; } bu;
      bu.h[0] = *(const v8h*)(bp);
      bu.h[1] = *(const v8h*)(bp + 8);
      acc[nt] = __builtin_amdgcn_wmma_f32_16x16x32_f16(
          false, au.v, false, bu.v, (short)0, acc[nt], false, false);
    }
  }

  // C/D layout: lanes 0-15 -> M = r, lanes 16-31 -> M = r + 8, N = ncol + 16*nt
  const int mbase = mtile * 16 + kgrp * 8;
#pragma unroll
  for (int nt = 0; nt < 8; ++nt) {
    const int n = nt * 16 + ncol;
    const float b = bias[n];
#pragma unroll
    for (int r = 0; r < 8; ++r) {
      const int mrow = mbase + r;
      if (mrow < nrows) {
        float val = acc[nt][r] + b;
        if (do_relu) val = fmaxf(val, 0.0f);
        out[(size_t)mrow * CH + n] = val;
      }
    }
  }
}

// ---------------------------------------------------------------- edge phase

__global__ void edge_score_kernel(const int* __restrict__ ei, const float* __restrict__ q,
                                  const float* __restrict__ k, float* __restrict__ esc,
                                  float* __restrict__ mx, int E)
{
  int idx = blockIdx.x * blockDim.x + threadIdx.x;
  if (idx >= E * HEADS) return;
  const int e   = idx >> 2;
  const int hh  = idx & (HEADS - 1);
  const int src = ei[e];
  const int dst = ei[E + e];
  const float* qp = q + (size_t)dst * CH + hh * HID;
  const float* kp = k + (size_t)src * CH + hh * HID;
  __builtin_prefetch(kp, 0, 0);              // global_prefetch_b8
  float s = 0.0f;
#pragma unroll
  for (int j = 0; j < HID; ++j) s = fmaf(qp[j], kp[j], s);
  s *= 0.17677669529663687f;                 // 1/sqrt(32)
  esc[idx] = s;
  atomicMaxF(&mx[(size_t)dst * HEADS + hh], s);
}

__global__ void edge_exp_kernel(const int* __restrict__ ei, const float* __restrict__ mx,
                                float* __restrict__ esc, float* __restrict__ ssum, int E)
{
  int idx = blockIdx.x * blockDim.x + threadIdx.x;
  if (idx >= E * HEADS) return;
  const int e  = idx >> 2;
  const int hh = idx & (HEADS - 1);
  const int dst = ei[E + e];
  float a = __expf(esc[idx] - mx[(size_t)dst * HEADS + hh]);
  esc[idx] = a;                              // overwrite score with exp value
  atomicAdd(&ssum[(size_t)dst * HEADS + hh], a);
}

__global__ void edge_aggr_kernel(const int* __restrict__ ei, const float* __restrict__ esc,
                                 const float* __restrict__ ssum, const float* __restrict__ v,
                                 float* __restrict__ agg, int E)
{
  int idx = blockIdx.x * blockDim.x + threadIdx.x;   // E*CH = 102.4M < 2^31
  if (idx >= E * CH) return;
  const int e   = idx >> 7;
  const int c   = idx & (CH - 1);
  const int hh  = c >> 5;
  const int src = ei[e];
  const int dst = ei[E + e];
  const float alpha = esc[(size_t)e * HEADS + hh] / ssum[(size_t)dst * HEADS + hh];
  atomicAdd(&agg[(size_t)dst * CH + c], alpha * v[(size_t)src * CH + c]);
}

__global__ void add_relu_kernel(float* __restrict__ dst, const float* __restrict__ a,
                                const float* __restrict__ b, int n)
{
  int i = blockIdx.x * blockDim.x + threadIdx.x;
  if (i < n) dst[i] = fmaxf(a[i] + b[i], 0.0f);
}

// ---------------------------------------------------------------- head + pool

__global__ void head_kernel(const float* __restrict__ h3, const float* __restrict__ wl2,
                            const float* __restrict__ bl2, float* __restrict__ hsc, int n)
{
  int i = blockIdx.x * blockDim.x + threadIdx.x;
  if (i >= n) return;
  const float* hp = h3 + (size_t)i * CH;
  float s = bl2[0];
#pragma unroll 8
  for (int j = 0; j < CH; ++j) s = fmaf(hp[j], wl2[j], s);
  hsc[i] = s;
}

__global__ void pool_kernel(const int* __restrict__ bat, const float* __restrict__ hsc,
                            float* __restrict__ sums, float* __restrict__ cnts, int n)
{
  int i = blockIdx.x * blockDim.x + threadIdx.x;
  if (i < n) {
    int g = bat[i];
    atomicAdd(&sums[g], hsc[i]);
    atomicAdd(&cnts[g], 1.0f);
  }
}

__global__ void pool_div_kernel(const float* __restrict__ sums, const float* __restrict__ cnts,
                                float* __restrict__ out, int ng)
{
  int g = blockIdx.x * blockDim.x + threadIdx.x;
  if (g < ng) out[g] = sums[g] / fmaxf(cnts[g], 1.0f);
}

// ---------------------------------------------------------------- driver

extern "C" void kernel_launch(void* const* d_in, const int* in_sizes, int n_in,
                              void* d_out, int out_size, void* d_ws, size_t ws_size,
                              hipStream_t stream)
{
  (void)n_in; (void)ws_size;
  const float* x   = (const float*)d_in[0];
  const int*   ei  = (const int*)  d_in[1];
  const int*   bat = (const int*)  d_in[2];
  const float* Wsrc[9] = { (const float*)d_in[3],  (const float*)d_in[5],
                           (const float*)d_in[7],  (const float*)d_in[9],
                           (const float*)d_in[11], (const float*)d_in[13],
                           (const float*)d_in[15], (const float*)d_in[17],
                           (const float*)d_in[19] };
  const float* bq1 = (const float*)d_in[4];
  const float* bk1 = (const float*)d_in[6];
  const float* bv1 = (const float*)d_in[8];
  const float* bs1 = (const float*)d_in[10];
  const float* bq2 = (const float*)d_in[12];
  const float* bk2 = (const float*)d_in[14];
  const float* bv2 = (const float*)d_in[16];
  const float* bs2 = (const float*)d_in[18];
  const float* bl  = (const float*)d_in[20];
  const float* wl2 = (const float*)d_in[21];
  const float* bl2 = (const float*)d_in[22];

  const int N  = in_sizes[0] / CH;     // 50000
  const int E  = in_sizes[1] / 2;      // 800000
  const int NG = out_size;             // 128

  // ---- workspace carve-up (~155 MB)
  char* wsb = (char*)d_ws;
  size_t off = 0;
  auto take = [&](size_t bytes) -> void* {
    void* p = wsb + off;
    off += (bytes + 255) & ~(size_t)255;
    return p;
  };
  const size_t NF = (size_t)N * CH;
  float*    q    = (float*)take(NF * 4);
  float*    k    = (float*)take(NF * 4);
  float*    v    = (float*)take(NF * 4);
  float*    skip = (float*)take(NF * 4);
  float*    h    = (float*)take(NF * 4);
  float*    esc  = (float*)take((size_t)E * HEADS * 4);
  float*    mx   = (float*)take((size_t)N * HEADS * 4);
  float*    ssum = (float*)take((size_t)N * HEADS * 4);
  float*    hsc  = (float*)take((size_t)N * 4);
  float*    pool = (float*)take(256 * 4);
  _Float16* xh   = (_Float16*)take(NF * 2);
  _Float16* wh   = (_Float16*)take((size_t)9 * CH * CH * 2);

  const int TB = 256;
  auto cdiv = [](long long a, long long b) { return (int)((a + b - 1) / b); };
  const int gNF = cdiv((long long)NF, TB);
  const int gW  = cdiv(CH * CH, TB);
  const int gMM = cdiv((N + 15) / 16, 8);
  const int gEH = cdiv((long long)E * HEADS, TB);
  const int gEC = cdiv((long long)E * CH, TB);
  const int gNH = cdiv((long long)N * HEADS, TB);
  const int gN  = cdiv(N, TB);

  // ---- f16 staging: activations plain, weights packed fragment-major
  f32_to_f16_kernel<<<gNF, TB, 0, stream>>>(x, xh, (int)NF);
  for (int i = 0; i < 9; ++i)
    pack_w_kernel<<<gW, TB, 0, stream>>>(Wsrc[i], wh + (size_t)i * CH * CH);
  _Float16* whq1 = wh + 0 * CH * CH; _Float16* whk1 = wh + 1 * CH * CH;
  _Float16* whv1 = wh + 2 * CH * CH; _Float16* whs1 = wh + 3 * CH * CH;
  _Float16* whq2 = wh + 4 * CH * CH; _Float16* whk2 = wh + 5 * CH * CH;
  _Float16* whv2 = wh + 6 * CH * CH; _Float16* whs2 = wh + 7 * CH * CH;
  _Float16* whl  = wh + 8 * CH * CH;

  // ================= Layer 1 =================
  wmma_gemm128<<<gMM, TB, 0, stream>>>(xh, whq1, bq1, q,    N, 0);
  wmma_gemm128<<<gMM, TB, 0, stream>>>(xh, whk1, bk1, k,    N, 0);
  wmma_gemm128<<<gMM, TB, 0, stream>>>(xh, whv1, bv1, v,    N, 0);
  wmma_gemm128<<<gMM, TB, 0, stream>>>(xh, whs1, bs1, skip, N, 0);

  fill_kernel<<<gNH, TB, 0, stream>>>(mx,   -3.0e38f, N * HEADS);
  fill_kernel<<<gNH, TB, 0, stream>>>(ssum,  0.0f,    N * HEADS);
  edge_score_kernel<<<gEH, TB, 0, stream>>>(ei, q, k, esc, mx, E);
  edge_exp_kernel  <<<gEH, TB, 0, stream>>>(ei, mx, esc, ssum, E);
  fill_kernel<<<gNF, TB, 0, stream>>>(h, 0.0f, (int)NF);
  edge_aggr_kernel <<<gEC, TB, 0, stream>>>(ei, esc, ssum, v, h, E);
  add_relu_kernel  <<<gNF, TB, 0, stream>>>(h, h, skip, (int)NF);   // h = relu(agg + skip)

  // ================= Layer 2 =================
  f32_to_f16_kernel<<<gNF, TB, 0, stream>>>(h, xh, (int)NF);
  wmma_gemm128<<<gMM, TB, 0, stream>>>(xh, whq2, bq2, q,    N, 0);
  wmma_gemm128<<<gMM, TB, 0, stream>>>(xh, whk2, bk2, k,    N, 0);
  wmma_gemm128<<<gMM, TB, 0, stream>>>(xh, whv2, bv2, v,    N, 0);
  wmma_gemm128<<<gMM, TB, 0, stream>>>(xh, whs2, bs2, skip, N, 0);

  fill_kernel<<<gNH, TB, 0, stream>>>(mx,   -3.0e38f, N * HEADS);
  fill_kernel<<<gNH, TB, 0, stream>>>(ssum,  0.0f,    N * HEADS);
  edge_score_kernel<<<gEH, TB, 0, stream>>>(ei, q, k, esc, mx, E);
  edge_exp_kernel  <<<gEH, TB, 0, stream>>>(ei, mx, esc, ssum, E);
  fill_kernel<<<gNF, TB, 0, stream>>>(k, 0.0f, (int)NF);            // reuse k as agg buffer
  edge_aggr_kernel <<<gEC, TB, 0, stream>>>(ei, esc, ssum, v, k, E);
  add_relu_kernel  <<<gNF, TB, 0, stream>>>(h, k, skip, (int)NF);   // h = relu(agg + skip)

  // ================= Head + pooling =================
  f32_to_f16_kernel<<<gNF, TB, 0, stream>>>(h, xh, (int)NF);
  wmma_gemm128<<<gMM, TB, 0, stream>>>(xh, whl, bl, q, N, 0);       // q = h @ wl + bl
  head_kernel<<<gN, TB, 0, stream>>>(q, wl2, bl2, hsc, N);          // hsc = q @ wl2 + bl2
  fill_kernel<<<1, 256, 0, stream>>>(pool, 0.0f, 256);
  pool_kernel<<<gN, TB, 0, stream>>>(bat, hsc, pool, pool + 128, N);
  pool_div_kernel<<<1, 256, 0, stream>>>(pool, pool + 128, (float*)d_out, NG);
}
